// EfficientMultiScaleAttention_19267223290221
// MI455X (gfx1250) — matrix operand order repaired and checked by reading, not compiled, not run on hardware
//
#include <hip/hip_runtime.h>
#include <hip/hip_bf16.h>
#include <cstdint>

// ---------------------------------------------------------------------------
// EfficientMultiScaleAttention for MI455X (gfx1250, wave32, WMMA).
// Heavy op: 3x3 conv (38.7 GFLOP) as implicit GEMM with
// v_wmma_f32_16x16x32_bf16 (bf16 inputs, f32 accumulate), weights staged in
// LDS via global_load_async_to_lds_b128, wide N-tiling (2 N-tiles x 4 M-tiles
// per wave -> 8 WMMAs per 6 ds_load pairs) + double-buffered fragments.
// ---------------------------------------------------------------------------

typedef __attribute__((ext_vector_type(16))) __bf16 v16bf;
typedef __attribute__((ext_vector_type(8)))  __bf16 v8bf;
typedef __attribute__((ext_vector_type(8)))  float  v8f;
typedef __attribute__((ext_vector_type(4)))  int    v4i;
typedef __attribute__((address_space(1))) v4i g_v4i;   // global int4
typedef __attribute__((address_space(3))) v4i l_v4i;   // LDS int4

#define NBG 128      // B*G
#define NGC 64       // channels per group
#define NH  64
#define NW  64
#define NHW 4096     // H*W
#define NCHW 262144  // GC*H*W

#if defined(__HIP_DEVICE_COMPILE__) && __has_builtin(__builtin_amdgcn_global_load_async_to_lds_b128)
#define HAVE_ASYNC_LDS 1
#else
#define HAVE_ASYNC_LDS 0
#endif

__device__ __forceinline__ float sigmoidf(float v) {
    return 1.0f / (1.0f + __expf(-v));
}

// --------------------------- kernel 0: row/col means -----------------------
__global__ void k_cat(const float* __restrict__ x, float* __restrict__ cat) {
    __shared__ float tile[64 * 65];
    const int t = threadIdx.x;
    const size_t base = (size_t)blockIdx.x * NHW;
    for (int k = 0; k < 16; ++k) {
        int p = t + k * 256;
        tile[(p >> 6) * 65 + (p & 63)] = x[base + p];
    }
    __syncthreads();
    if (t < 64) {
        float s = 0.f;
        #pragma unroll 8
        for (int j = 0; j < 64; ++j) s += tile[t * 65 + j];
        cat[(size_t)blockIdx.x * 128 + t] = s * (1.0f / 64.0f);
    } else if (t < 128) {
        const int u = t - 64;
        float s = 0.f;
        #pragma unroll 8
        for (int r = 0; r < 64; ++r) s += tile[r * 65 + u];
        cat[(size_t)blockIdx.x * 128 + 64 + u] = s * (1.0f / 64.0f);
    }
}

// --------------------------- kernel 1: w1 GEMM + sigmoid -------------------
__global__ void k_gate(const float* __restrict__ w1, const float* __restrict__ cat,
                       float* __restrict__ sigH, float* __restrict__ sigW) {
    __shared__ float w1s[64 * 64];
    __shared__ float cs[64 * 129];
    const int bg = blockIdx.x, t = threadIdx.x;
    for (int k = 0; k < 16; ++k) w1s[t + k * 256] = w1[t + k * 256];
    for (int k = 0; k < 32; ++k) {
        int p = t + k * 256;                 // 0..8191 over 64x128
        cs[(p >> 7) * 129 + (p & 127)] = cat[(size_t)bg * 8192 + p];
    }
    __syncthreads();
    for (int k = 0; k < 32; ++k) {
        int p = t + k * 256;
        int o = p >> 7, q = p & 127;
        float s = 0.f;
        #pragma unroll 8
        for (int i = 0; i < 64; ++i) s += w1s[o * 64 + i] * cs[i * 129 + q];
        float g = sigmoidf(s);
        if (q < 64) sigH[((size_t)bg * 64 + o) * 64 + q] = g;
        else        sigW[((size_t)bg * 64 + o) * 64 + (q - 64)] = g;
    }
}

// ------------------- kernel 2: x1p + per-channel stats ---------------------
__global__ void k_x1p(const float* __restrict__ x, const float* __restrict__ sigH,
                      const float* __restrict__ sigW, float* __restrict__ x1p,
                      float* __restrict__ chsum, float* __restrict__ chsumsq) {
    __shared__ float sh[64], sw[64];
    __shared__ float rs[256], rq[256];
    const int t = threadIdx.x;
    const size_t bgc = blockIdx.x;           // bg*64 + c
    const size_t base = bgc * NHW;
    if (t < 64)       sh[t]      = sigH[bgc * 64 + t];
    else if (t < 128) sw[t - 64] = sigW[bgc * 64 + (t - 64)];
    __syncthreads();
    float ls = 0.f, lq = 0.f;
    for (int k = 0; k < 16; ++k) {
        int p = t + k * 256;
        float v = x[base + p] * sh[p >> 6] * sw[p & 63];
        x1p[base + p] = v;
        ls += v; lq += v * v;
    }
    rs[t] = ls; rq[t] = lq;
    __syncthreads();
    for (int s = 128; s > 0; s >>= 1) {
        if (t < s) { rs[t] += rs[t + s]; rq[t] += rq[t + s]; }
        __syncthreads();
    }
    if (t == 0) { chsum[blockIdx.x] = rs[0]; chsumsq[blockIdx.x] = rq[0]; }
}

// ------------- kernel W: swizzle w3 into WMMA A-fragment layout ------------
// wB[mt][tap][chunk][lane][j] per ISA 7.12.2 16-bit A 16x32 layout:
//   M = mt*16 + lane%16 ; K = (j/8)*16 + (lane/16)*8 + (j%8) ; i = chunk*32 + K
__global__ void k_wprep(const float* __restrict__ w3, __bf16* __restrict__ wB) {
    const int idx = blockIdx.x * 256 + threadIdx.x;   // 36864 total
    const int j     = idx & 15;
    const int L     = (idx >> 4) & 31;
    const int chunk = (idx >> 9) & 1;
    const int tap   = (idx >> 10) % 9;
    const int mt    = idx / 9216;
    const int o = mt * 16 + (L & 15);
    const int K = ((j >> 3) << 4) + ((L >> 4) << 3) + (j & 7);
    const int i = chunk * 32 + K;
    const int ky = tap / 3, kx = tap % 3;
    wB[idx] = (__bf16)w3[((o * 64 + i) * 3 + ky) * 3 + kx];
}

// ---------------- kernel C: 3x3 conv via bf16 WMMA implicit GEMM -----------
// block = (bg, ystrip of 4 rows); 8 waves; wave -> one row x two 16-col
// N-tiles x all 4 M-tiles = 8 accumulators; 18 K-steps -> 144 WMMAs/wave.
// Per K-step: 2 B-frag pairs + 4 A-frag pairs feed 8 WMMAs.
#define LDS_CSTRIDE 72   // padded channel stride (16B-aligned frags, no bank conflicts)

__global__ void __launch_bounds__(256) k_conv(const float* __restrict__ x,
                                              const __bf16* __restrict__ wB,
                                              float* __restrict__ x2) {
    __shared__ __align__(128) __bf16 xsb[6 * 66 * LDS_CSTRIDE];   // 57024 B
    __shared__ __align__(128) __bf16 wsb[36864];                  // 73728 B
    const int bg = blockIdx.x;
    const int y0 = blockIdx.y * 4;
    const int t = threadIdx.x;

    // ---- stage pre-swizzled weights into LDS (async DMA path if available) ----
#if HAVE_ASYNC_LDS
    {
        g_v4i* gsrc = (g_v4i*)((const char*)wB + t * 16);
        l_v4i* ldst = (l_v4i*)((char*)wsb + t * 16);
        #pragma unroll
        for (int i = 0; i < 18; ++i)
            __builtin_amdgcn_global_load_async_to_lds_b128(gsrc + i * 256, ldst + i * 256,
                                                           0, 0);
    }
#else
    {
        const uint4* gs = (const uint4*)wB;
        uint4* ld = (uint4*)wsb;
        #pragma unroll
        for (int i = 0; i < 18; ++i) ld[t + i * 256] = gs[t + i * 256];
    }
#endif

    // stage input tile (6 rows incl. halo, 66 cols incl. halo, 64 ch) as bf16
    for (int slot = t; slot < 384; slot += 256) {
        const int r = slot >> 6, col = slot & 63;
        const int gy = y0 - 1 + r;
        if (gy >= 0 && gy < NH) {
            const float* xp = x + (size_t)bg * NCHW + (size_t)gy * NW + col;
            #pragma unroll 4
            for (int ch = 0; ch < 64; ++ch)
                xsb[((r * 66) + col + 1) * LDS_CSTRIDE + ch] = (__bf16)xp[(size_t)ch * NHW];
        } else {
            #pragma unroll 4
            for (int ch = 0; ch < 64; ++ch)
                xsb[((r * 66) + col + 1) * LDS_CSTRIDE + ch] = (__bf16)0.0f;
        }
    }
    // zero halo columns 0 and 65 (col index reused as channel index)
    for (int slot = t; slot < 384; slot += 256) {
        const int r = slot >> 6, ch = slot & 63;
        xsb[(r * 66 + 0)  * LDS_CSTRIDE + ch] = (__bf16)0.0f;
        xsb[(r * 66 + 65) * LDS_CSTRIDE + ch] = (__bf16)0.0f;
    }

#if HAVE_ASYNC_LDS
#if __has_builtin(__builtin_amdgcn_s_wait_asynccnt)
    __builtin_amdgcn_s_wait_asynccnt(0);
#else
    asm volatile("s_wait_asynccnt 0x0" ::: "memory");
#endif
#endif
    __syncthreads();

    const int lane = t & 31, wv = t >> 5;
    const int ylocal = wv >> 1;              // 0..3 (output row in strip)
    const int xb0 = (wv & 1) << 5;           // 0 or 32; tiles at xb0, xb0+16
    const int hh = lane >> 4, n = lane & 15;

    v8f acc[4][2] = {};

    auto loadB = [&](int s, int tl) -> v16bf {   // s = tap*2 + chunk
        const int tap = s >> 1, chunk = s & 1;
        const int row  = ylocal + tap / 3;
        const int colb = xb0 + tl * 16 + n + tap % 3;
        const __bf16* bp = &xsb[(row * 66 + colb) * LDS_CSTRIDE + chunk * 32 + hh * 8];
        v8bf lo = *(const v8bf*)(bp);
        v8bf hi = *(const v8bf*)(bp + 16);
        return __builtin_shufflevector(lo, hi,
            0, 1, 2, 3, 4, 5, 6, 7, 8, 9, 10, 11, 12, 13, 14, 15);
    };
    auto loadA = [&](int s, int mt) -> v16bf {
        return *(const v16bf*)&wsb[mt * 9216 + s * 512 + lane * 16];
    };

    // double-buffered main loop: fragments for step s+1 in flight while step
    // s's 8 WMMAs execute.
    v16bf b0 = loadB(0, 0), b1 = loadB(0, 1);
    v16bf a0 = loadA(0, 0), a1 = loadA(0, 1), a2 = loadA(0, 2), a3 = loadA(0, 3);
    #pragma unroll
    for (int s = 0; s < 18; ++s) {
        v16bf nb0 = b0, nb1 = b1, n0 = a0, n1 = a1, n2 = a2, n3 = a3;
        if (s < 17) {
            nb0 = loadB(s + 1, 0);
            nb1 = loadB(s + 1, 1);
            n0 = loadA(s + 1, 0);
            n1 = loadA(s + 1, 1);
            n2 = loadA(s + 1, 2);
            n3 = loadA(s + 1, 3);
        }
        acc[0][0] = __builtin_amdgcn_wmma_f32_16x16x32_bf16(false, a0, false, b0, (short)0, acc[0][0], false, false);
        acc[0][1] = __builtin_amdgcn_wmma_f32_16x16x32_bf16(false, a0, false, b1, (short)0, acc[0][1], false, false);
        acc[1][0] = __builtin_amdgcn_wmma_f32_16x16x32_bf16(false, a1, false, b0, (short)0, acc[1][0], false, false);
        acc[1][1] = __builtin_amdgcn_wmma_f32_16x16x32_bf16(false, a1, false, b1, (short)0, acc[1][1], false, false);
        acc[2][0] = __builtin_amdgcn_wmma_f32_16x16x32_bf16(false, a2, false, b0, (short)0, acc[2][0], false, false);
        acc[2][1] = __builtin_amdgcn_wmma_f32_16x16x32_bf16(false, a2, false, b1, (short)0, acc[2][1], false, false);
        acc[3][0] = __builtin_amdgcn_wmma_f32_16x16x32_bf16(false, a3, false, b0, (short)0, acc[3][0], false, false);
        acc[3][1] = __builtin_amdgcn_wmma_f32_16x16x32_bf16(false, a3, false, b1, (short)0, acc[3][1], false, false);
        b0 = nb0; b1 = nb1; a0 = n0; a1 = n1; a2 = n2; a3 = n3;
    }

    // D layout: lane -> N=n ; vgpr r -> M=r+hh*8
    const int y = y0 + ylocal;
    #pragma unroll
    for (int mt = 0; mt < 4; ++mt) {
        #pragma unroll
        for (int tl = 0; tl < 2; ++tl) {
            #pragma unroll
            for (int r = 0; r < 8; ++r) {
                const int o = mt * 16 + hh * 8 + r;
                x2[((size_t)bg * 64 + o) * NHW + (size_t)y * NW + xb0 + tl * 16 + n] =
                    acc[mt][tl][r];
            }
        }
    }
}

// ---------------- kernel S: per-channel sum of a [bg][c][hw] tensor --------
__global__ void k_chansum(const float* __restrict__ src, float* __restrict__ dst) {
    __shared__ float rs[256];
    const int t = threadIdx.x;
    const size_t base = (size_t)blockIdx.x * NHW;
    float ls = 0.f;
    for (int k = 0; k < 16; ++k) ls += src[base + t + k * 256];
    rs[t] = ls;
    __syncthreads();
    for (int s = 128; s > 0; s >>= 1) {
        if (t < s) rs[t] += rs[t + s];
        __syncthreads();
    }
    if (t == 0) dst[blockIdx.x] = rs[0];
}

// ---------------- kernel F: GN stats + dual softmax (per bg) ---------------
__global__ void k_softmax(const float* __restrict__ chsum1, const float* __restrict__ chsumsq1,
                          const float* __restrict__ chsum2,
                          const float* __restrict__ gamma, const float* __restrict__ beta,
                          float* __restrict__ s1, float* __restrict__ s2,
                          float* __restrict__ muArr, float* __restrict__ invArr) {
    __shared__ float ra[64], rb[64];
    const int bg = blockIdx.x, c = threadIdx.x;
    const float su = chsum1[bg * 64 + c], sq = chsumsq1[bg * 64 + c];
    ra[c] = su; rb[c] = sq;
    __syncthreads();
    for (int s = 32; s > 0; s >>= 1) {
        if (c < s) { ra[c] += ra[c + s]; rb[c] += rb[c + s]; }
        __syncthreads();
    }
    const float S = ra[0], Q = rb[0];
    __syncthreads();
    const float mu  = S * (1.0f / 262144.0f);
    const float var = Q * (1.0f / 262144.0f) - mu * mu;
    const float inv = rsqrtf(var + 1e-5f);
    const float m1 = (su * (1.0f / 4096.0f) - mu) * inv * gamma[c] + beta[c];
    const float m2 = chsum2[bg * 64 + c] * (1.0f / 4096.0f);
    ra[c] = m1; rb[c] = m2;
    __syncthreads();
    for (int s = 32; s > 0; s >>= 1) {
        if (c < s) { ra[c] = fmaxf(ra[c], ra[c + s]); rb[c] = fmaxf(rb[c], rb[c + s]); }
        __syncthreads();
    }
    const float M1 = ra[0], M2 = rb[0];
    __syncthreads();
    const float e1 = __expf(m1 - M1), e2 = __expf(m2 - M2);
    ra[c] = e1; rb[c] = e2;
    __syncthreads();
    for (int s = 32; s > 0; s >>= 1) {
        if (c < s) { ra[c] += ra[c + s]; rb[c] += rb[c + s]; }
        __syncthreads();
    }
    s1[bg * 64 + c] = e1 / ra[0];
    s2[bg * 64 + c] = e2 / rb[0];
    if (c == 0) { muArr[bg] = mu; invArr[bg] = inv; }
}

// ---------------- kernel M: wmap = s1.x2 + s2.x1 (channel contraction) -----
__global__ void k_wmap(const float* __restrict__ x1p, const float* __restrict__ x2,
                       const float* __restrict__ s1, const float* __restrict__ s2,
                       const float* __restrict__ gamma, const float* __restrict__ beta,
                       const float* __restrict__ muArr, const float* __restrict__ invArr,
                       float* __restrict__ wmap) {
    __shared__ float s1s[64], s2s[64], gs[64], bs[64];
    const int bg = blockIdx.x;
    const int t = threadIdx.x;
    const int pos = blockIdx.y * 256 + t;
    if (t < 64) {
        s1s[t] = s1[bg * 64 + t];
        s2s[t] = s2[bg * 64 + t];
        gs[t]  = gamma[t];
        bs[t]  = beta[t];
    }
    __syncthreads();
    const float mu = muArr[bg], inv = invArr[bg];
    float acc = 0.f;
    const size_t base = (size_t)bg * NCHW + pos;
    for (int c = 0; c < 64; ++c) {
        const size_t o = base + (size_t)c * NHW;
        __builtin_prefetch(&x2[o + NHW], 0, 0);     // global_prefetch_b8
        __builtin_prefetch(&x1p[o + NHW], 0, 0);
        const float a2 = x2[o];
        const float a1 = (x1p[o] - mu) * inv * gs[c] + bs[c];
        acc += s1s[c] * a2 + s2s[c] * a1;
    }
    wmap[(size_t)bg * NHW + pos] = acc;
}

// ---------------- kernel O: out = x * sigmoid(wmap) ------------------------
__global__ void k_out(const float* __restrict__ x, const float* __restrict__ wmap,
                      float* __restrict__ out) {
    const size_t idx = (size_t)blockIdx.x * 256 + threadIdx.x;
    const int bg  = (int)(idx >> 18);        // / (64*4096)
    const int pos = (int)(idx & 4095);
    out[idx] = x[idx] * sigmoidf(wmap[(size_t)bg * NHW + pos]);
}

// ---------------------------------------------------------------------------
extern "C" void kernel_launch(void* const* d_in, const int* in_sizes, int n_in,
                              void* d_out, int out_size, void* d_ws, size_t ws_size,
                              hipStream_t stream) {
    const float* x     = (const float*)d_in[0];
    const float* w1    = (const float*)d_in[1];
    const float* w3    = (const float*)d_in[2];
    const float* gamma = (const float*)d_in[3];
    const float* beta  = (const float*)d_in[4];
    float* out = (float*)d_out;

    char* ws = (char*)d_ws;
    size_t off = 0;
    auto take = [&](size_t bytes) -> char* {
        char* p = ws + off;
        off = (off + bytes + 255) & ~(size_t)255;
        return p;
    };
    float*  cat      = (float*)take((size_t)NBG * 64 * 128 * 4);   //   4 MB
    float*  sigH     = (float*)take((size_t)NBG * 64 * 64 * 4);    //   2 MB
    float*  sigW     = (float*)take((size_t)NBG * 64 * 64 * 4);    //   2 MB
    float*  x1p      = (float*)take((size_t)NBG * NCHW * 4);       // 128 MB
    float*  chsum1   = (float*)take(8192 * 4);
    float*  chsumsq1 = (float*)take(8192 * 4);
    float*  chsum2   = (float*)take(8192 * 4);
    float*  s1       = (float*)take(8192 * 4);
    float*  s2       = (float*)take(8192 * 4);
    float*  muArr    = (float*)take(128 * 4);
    float*  invArr   = (float*)take(128 * 4);
    float*  wmapB    = (float*)take((size_t)NBG * NHW * 4);        //   2 MB
    __bf16* wB       = (__bf16*)take(36864 * 2);

    float* x2 = out;   // stage conv output in d_out; fully overwritten by k_out

    k_wprep  <<<dim3(144),       dim3(256), 0, stream>>>(w3, wB);
    k_cat    <<<dim3(8192),      dim3(256), 0, stream>>>(x, cat);
    k_gate   <<<dim3(128),       dim3(256), 0, stream>>>(w1, cat, sigH, sigW);
    k_x1p    <<<dim3(8192),      dim3(256), 0, stream>>>(x, sigH, sigW, x1p, chsum1, chsumsq1);
    k_conv   <<<dim3(128, 16),   dim3(256), 0, stream>>>(x, wB, x2);
    k_chansum<<<dim3(8192),      dim3(256), 0, stream>>>(x2, chsum2);
    k_softmax<<<dim3(128),       dim3(64),  0, stream>>>(chsum1, chsumsq1, chsum2, gamma, beta,
                                                         s1, s2, muArr, invArr);
    k_wmap   <<<dim3(128, 16),   dim3(256), 0, stream>>>(x1p, x2, s1, s2, gamma, beta,
                                                         muArr, invArr, wmapB);
    k_out    <<<dim3(131072),    dim3(256), 0, stream>>>(x, wmapB, out);

    (void)in_sizes; (void)n_in; (void)out_size; (void)ws_size;
}
